// Attention_50457275793495
// MI455X (gfx1250) — matrix operand order repaired and checked by reading, not compiled, not run on hardware
//
#include <hip/hip_runtime.h>
#include <hip/hip_bf16.h>
#include <math.h>

typedef __attribute__((ext_vector_type(16))) _Float16 v16h;
typedef __attribute__((ext_vector_type(8)))  _Float16 v8h;
typedef __attribute__((ext_vector_type(8)))  float    v8f;
typedef __attribute__((ext_vector_type(4)))  unsigned int u32x4;
typedef __attribute__((ext_vector_type(8)))  int      i32x8;
typedef __attribute__((ext_vector_type(4)))  int      i32x4;

#define B_ 2
#define L_ 2048
#define D_ 256
#define H_ 8
#define C_ 32
#define O_ 256
#define QSCALE 0.17677669529663687f   // C^-0.5

// ---------------------------------------------------------------------------
// Prep 0a: transpose f32 weight [K][N] -> f16 [N][K] so WMMA-B operands are
// contiguous 32B per lane. scaleMode=1 pre-scales q columns ((n&63)<32) by
// C^-0.5, folding the softmax scale into the weights.
// ---------------------------------------------------------------------------
__global__ void prepW_kernel(const float* __restrict__ src,
                             _Float16* __restrict__ dst,
                             int K, int N, int scaleMode)
{
    int tid = blockIdx.x * 256 + threadIdx.x;
    if (tid >= K * N) return;
    int k = tid % K, n = tid / K;
    float v = src[(size_t)k * N + n];
    if (scaleMode && ((n & 63) < 32)) v *= QSCALE;
    dst[(size_t)n * K + k] = (_Float16)v;
}

// Prep 0b: pre-scaled qg bias (q half scaled by C^-0.5, gate half unscaled).
__global__ void prepB_kernel(const float* __restrict__ qgB,
                             float* __restrict__ qgBs)
{
    int o = blockIdx.x * 256 + threadIdx.x;
    if (o < 512) {
        float v = qgB[o];
        if ((o & 63) < 32) v *= QSCALE;
        qgBs[o] = v;
    }
}

// ---------------------------------------------------------------------------
// Kernel 1: qg / kv projections on the matrix pipe.
// One block = (b, 16-row l tile), 4 waves; input rows staged in LDS as f16.
// K=256 = 8 WMMA chunks, two independent accumulation chains per N tile.
// ---------------------------------------------------------------------------
__global__ void projW_kernel(const float* __restrict__ qin,
                             const float* __restrict__ kvin,
                             const _Float16* __restrict__ qgWt,  // [512][256]
                             const _Float16* __restrict__ kvWt,  // [512][256]
                             const float* __restrict__ qgBs,
                             const float* __restrict__ kvB,
                             _Float16* __restrict__ qf,
                             _Float16* __restrict__ kf,
                             _Float16* __restrict__ vT,
                             float* __restrict__ g32)
{
    __shared__ _Float16 rows[2][16 * D_];
    int blk = blockIdx.x;
    int b = blk >> 7;
    int q0 = (blk & 127) * 16;
    int t = threadIdx.x;                     // 128 threads = 4 waves
    int lane = t & 31, w = t >> 5;
    int m = lane & 15, hi = lane >> 4;

    for (int idx = t; idx < 16 * D_; idx += 128) {
        int li = idx >> 8, d = idx & 255;
        rows[0][idx] = (_Float16)qin [((size_t)(b * L_ + q0 + li)) * D_ + d];
        rows[1][idx] = (_Float16)kvin[((size_t)(b * L_ + q0 + li)) * D_ + d];
    }
    __syncthreads();

    for (int j = w; j < 64; j += 4) {
        int which = j & 1;
        int n0 = (j >> 1) * 16;
        const _Float16* Wt = which ? kvWt : qgWt;
        const _Float16* rowsA = rows[which];
        v8f ca = {}, cb = {};
#pragma unroll
        for (int kcp = 0; kcp < 4; ++kcp) {
            int kc0 = 2 * kcp, kc1 = 2 * kcp + 1;
            const _Float16* a0p = rowsA + m * D_ + kc0 * 32 + hi * 16;
            const _Float16* a1p = rowsA + m * D_ + kc1 * 32 + hi * 16;
            v8h a0l = *(const v8h*)a0p, a0h = *(const v8h*)(a0p + 8);
            v8h a1l = *(const v8h*)a1p, a1h = *(const v8h*)(a1p + 8);
            v16h a0, a1;
#pragma unroll
            for (int i = 0; i < 8; ++i) {
                a0[i] = a0l[i]; a0[i + 8] = a0h[i];
                a1[i] = a1l[i]; a1[i + 8] = a1h[i];
            }
            v16h b0 = *(const v16h*)(Wt + (size_t)(n0 + m) * D_ + kc0 * 32 + hi * 16);
            v16h b1 = *(const v16h*)(Wt + (size_t)(n0 + m) * D_ + kc1 * 32 + hi * 16);
            ca = __builtin_amdgcn_wmma_f32_16x16x32_f16(false, a0, false, b0,
                                                        (short)0, ca, false, false);
            cb = __builtin_amdgcn_wmma_f32_16x16x32_f16(false, a1, false, b1,
                                                        (short)0, cb, false, false);
        }
        v8f acc = ca + cb;
        int o = n0 + m;
        int h = o >> 6, cc = o & 63;
        float bv = which ? kvB[o] : qgBs[o];
        size_t bh = (size_t)(b * H_ + h);
#pragma unroll
        for (int r = 0; r < 8; ++r) {
            int l = q0 + r + hi * 8;
            float val = acc[r] + bv;
            if (which == 0) {
                if (cc < C_) qf[(bh * L_ + l) * C_ + cc] = (_Float16)val;
                else         g32[(bh * L_ + l) * C_ + (cc - C_)] = val;
            } else {
                if (cc < C_) kf[(bh * L_ + l) * C_ + cc] = (_Float16)val;
                else         vT[(bh * C_ + (cc - C_)) * L_ + l] = (_Float16)val;
            }
        }
    }
}

// ---------------------------------------------------------------------------
// Kernel 2: fused attention. One block = (b, h, 16-row q block), 4 waves.
// The 16x2048 bias tile is streamed into LDS by ONE Tensor-Data-Mover
// descriptor issued by wave 0 (tracked by TENSORcnt), fully overlapped with
// the Q.K^T WMMA phase; softmax then adds bias from LDS.
// h is the fastest grid dim so the 8 head blocks of one (b,qt) interleave
// their (…,Lk,H) attn stores and combine into full lines in L2.
// LDS: 128KB logits + 128KB bias + 1KB scratch = 257KB (<= 320KB WGP LDS).
// ---------------------------------------------------------------------------
__global__ void attn_kernel(const _Float16* __restrict__ qf,
                            const _Float16* __restrict__ kf,
                            const _Float16* __restrict__ vT,
                            const float* __restrict__ g32,
                            const float* __restrict__ bias,
                            float* __restrict__ attn_out,     // (B,Lq,Lk,H)
                            _Float16* __restrict__ outg16)    // (B,H,L,C) f16
{
    extern __shared__ float lg[];            // [16][2048] logits
    float* red = lg + 16 * L_;               // 256 f32 scratch
    float* bls = lg + 16 * L_ + 256;         // [16][2048] bias tile (TDM dest)

    int blk = blockIdx.x;
    int h  = blk & 7;
    int qt = (blk >> 3) & 127;
    int b  = blk >> 10;
    int q0 = qt * 16;

    int t = threadIdx.x;                     // 128 threads = 4 waves
    int lane = t & 31, w = t >> 5;
    int m = lane & 15, hi = lane >> 4;
    size_t bh = (size_t)(b * H_ + h);

    // ---- kick off TDM: bias[b][h][q0..q0+15][0..2047] -> LDS (async) ----
    if (w == 0) {
        const float* gsrc = bias + (bh * L_ + q0) * L_;
        unsigned long long ga = (unsigned long long)(size_t)gsrc;
        unsigned int ldsoff = (unsigned int)(size_t)bls;   // LDS byte offset
        u32x4 g0;
        g0[0] = 1u;                                        // count=1, user D#
        g0[1] = ldsoff;                                    // lds_addr
        g0[2] = (unsigned int)ga;                          // global_addr[31:0]
        g0[3] = (unsigned int)((ga >> 32) & 0x1FFFFFFu)    // global_addr[56:32]
              | (2u << 30);                                // type=2 (image)
        unsigned int td0 = L_;     // tensor_dim0 (row length, elements)
        unsigned int td1 = 16u;    // tensor_dim1 (rows)
        i32x8 g1;
        g1[0] = (int)(2u << 16);                           // data_size=4B
        g1[1] = (int)((td0 & 0xFFFFu) << 16);              // tensor_dim0[15:0]
        g1[2] = (int)(((td0 >> 16) & 0xFFFFu)              // tensor_dim0[31:16]
              |       ((td1 & 0xFFFFu) << 16));            // tensor_dim1[15:0]
        g1[3] = (int)(((td1 >> 16) & 0xFFFFu)              // tensor_dim1[31:16]
              |       ((unsigned int)L_ << 16));           // tile_dim0 = 2048
        g1[4] = 16;                                        // tile_dim1=16, tile_dim2=0
        g1[5] = (int)L_;                                   // dim0_stride[31:0] = 2048
        g1[6] = 0;                                         // dim0_stride[47:32], dim1_stride lo
        g1[7] = 0;
        i32x4 z4 = {0, 0, 0, 0};
#if __clang_major__ >= 23
        i32x8 z8 = {0, 0, 0, 0, 0, 0, 0, 0};
        __builtin_amdgcn_tensor_load_to_lds(g0, g1, z4, z4, z8, 0);
#else
        __builtin_amdgcn_tensor_load_to_lds(g0, g1, z4, z4, 0);
#endif
    }

    // ---- A operand: Q tile (16x32 f16), pre-scaled by C^-0.5 ----
    const _Float16* qbase = qf + (bh * L_ + q0 + m) * C_ + hi * 16;
    v8h qlo = *(const v8h*)(qbase);
    v8h qhi = *(const v8h*)(qbase + 8);
    v16h aQ;
#pragma unroll
    for (int i = 0; i < 8; ++i) { aQ[i] = qlo[i]; aQ[i + 8] = qhi[i]; }

    // ---- Phase 1: raw logits = Q K^T (bias arrives via TDM meanwhile) ----
    for (int kt = w * 2; kt < L_ / 16; kt += 8) {
        const _Float16* kb0 = kf + (bh * L_ + kt * 16 + m) * C_ + hi * 16;
        const _Float16* kb1 = kf + (bh * L_ + (kt + 1) * 16 + m) * C_ + hi * 16;
        v16h bK0 = *(const v16h*)kb0;
        v16h bK1 = *(const v16h*)kb1;
        v8f acc0 = {}, acc1 = {};
        acc0 = __builtin_amdgcn_wmma_f32_16x16x32_f16(false, aQ, false, bK0,
                                                      (short)0, acc0, false, false);
        acc1 = __builtin_amdgcn_wmma_f32_16x16x32_f16(false, aQ, false, bK1,
                                                      (short)0, acc1, false, false);
        int col0 = kt * 16 + m;
#pragma unroll
        for (int r = 0; r < 8; ++r) {
            int row = r + hi * 8;
            lg[row * L_ + col0]      = acc0[r];
            lg[row * L_ + col0 + 16] = acc1[r];
        }
    }
    __syncthreads();
    if (w == 0) __builtin_amdgcn_s_wait_tensorcnt(0);   // bias tile landed
    __syncthreads();

    // ---- Phase 2: add bias (from LDS) + row softmax (8 threads/row) ----
    int r = t >> 3, s = t & 7;
    float* rowp = lg + r * L_;
    const float* brow = bls + r * L_;
    float lmax = -1e30f;
    for (int j = s * 256; j < s * 256 + 256; ++j) {
        float v = rowp[j] + brow[j];
        rowp[j] = v;
        lmax = fmaxf(lmax, v);
    }
    red[t] = lmax;
    __syncthreads();
    if (t < 16) {
        float mx = red[t * 8];
        for (int j = 1; j < 8; ++j) mx = fmaxf(mx, red[t * 8 + j]);
        red[128 + t] = mx;
    }
    __syncthreads();
    float rmax = red[128 + r];
    float lsum = 0.f;
    for (int j = s * 256; j < s * 256 + 256; ++j) {
        float e = __expf(rowp[j] - rmax);
        rowp[j] = e;
        lsum += e;
    }
    red[t] = lsum;
    __syncthreads();
    if (t < 16) {
        float sm = 0.f;
        for (int j = 0; j < 8; ++j) sm += red[t * 8 + j];
        red[144 + t] = sm;
    }
    __syncthreads();
    float inv = 1.f / red[144 + r];
    float* aout = attn_out + ((size_t)(b * L_ + q0 + r) * L_) * H_ + h;
    for (int j = s * 256; j < s * 256 + 256; ++j) {
        float p = rowp[j] * inv;
        rowp[j] = p;
        aout[(size_t)j * H_] = p;
    }
    __syncthreads();

    // ---- Phase 3: P @ V, two independent chains per accumulator ----
    v8f a0c0 = {}, a0c1 = {}, a1c0 = {}, a1c1 = {};
    for (int kc = w; kc < L_ / 32; kc += 8) {
        int kc2 = kc + 4;
        const float* p0 = lg + m * L_ + kc  * 32 + hi * 16;
        const float* p1 = lg + m * L_ + kc2 * 32 + hi * 16;
        v16h aP0, aP1;
#pragma unroll
        for (int i = 0; i < 8; ++i) {
            aP0[i] = (_Float16)p0[i]; aP0[i + 8] = (_Float16)p0[i + 8];
            aP1[i] = (_Float16)p1[i]; aP1[i + 8] = (_Float16)p1[i + 8];
        }
        v16h bV00 = *(const v16h*)(vT + (bh * C_ + m)      * L_ + kc  * 32 + hi * 16);
        v16h bV01 = *(const v16h*)(vT + (bh * C_ + 16 + m) * L_ + kc  * 32 + hi * 16);
        v16h bV10 = *(const v16h*)(vT + (bh * C_ + m)      * L_ + kc2 * 32 + hi * 16);
        v16h bV11 = *(const v16h*)(vT + (bh * C_ + 16 + m) * L_ + kc2 * 32 + hi * 16);
        a0c0 = __builtin_amdgcn_wmma_f32_16x16x32_f16(false, aP0, false, bV00,
                                                      (short)0, a0c0, false, false);
        a0c1 = __builtin_amdgcn_wmma_f32_16x16x32_f16(false, aP0, false, bV01,
                                                      (short)0, a0c1, false, false);
        a1c0 = __builtin_amdgcn_wmma_f32_16x16x32_f16(false, aP1, false, bV10,
                                                      (short)0, a1c0, false, false);
        a1c1 = __builtin_amdgcn_wmma_f32_16x16x32_f16(false, aP1, false, bV11,
                                                      (short)0, a1c1, false, false);
    }
    v8f acc0 = a0c0 + a1c0;
    v8f acc1 = a0c1 + a1c1;
    __syncthreads();                         // done reading lg; reuse as scratch

#pragma unroll
    for (int r2 = 0; r2 < 8; ++r2) {
        int mo = r2 + hi * 8;
        lg[w * 512 + mo * 32 + m]      = acc0[r2];
        lg[w * 512 + mo * 32 + 16 + m] = acc1[r2];
    }
    __syncthreads();

    for (int i = t * 4; i < t * 4 + 4; ++i) {
        int mo = i >> 5, c = i & 31;
        float sum = lg[i] + lg[512 + i] + lg[1024 + i] + lg[1536 + i];
        float gv = g32[(bh * L_ + q0 + mo) * C_ + c];
        float gate = 1.f / (1.f + __expf(-gv));
        outg16[(bh * L_ + q0 + mo) * C_ + c] = (_Float16)(sum * gate);
    }
}

// ---------------------------------------------------------------------------
// Kernel 3: output projection on the matrix pipe (K = H*C = 256).
// ---------------------------------------------------------------------------
__global__ void outW_kernel(const _Float16* __restrict__ outg16,
                            const _Float16* __restrict__ oWt,  // [256][256]
                            const float* __restrict__ oB,      // (O,1)
                            float* __restrict__ out)           // (B,L,O)
{
    __shared__ _Float16 arow[16 * 256];
    int blk = blockIdx.x;
    int b = blk >> 7;
    int q0 = (blk & 127) * 16;
    int t = threadIdx.x;
    int lane = t & 31, w = t >> 5;
    int m = lane & 15, hi = lane >> 4;

    {
        int h = t >> 4, mr = t & 15;
        const _Float16* src = outg16 + ((size_t)(b * H_ + h) * L_ + q0 + mr) * C_;
        _Float16* dst = arow + (h * 16 + mr) * 32;
        *(v16h*)(dst)      = *(const v16h*)(src);
        *(v16h*)(dst + 16) = *(const v16h*)(src + 16);
    }
    __syncthreads();

    for (int nt = w; nt < 16; nt += 4) {
        int n0 = nt * 16;
        v8f ca = {}, cb = {};
#pragma unroll
        for (int kcp = 0; kcp < 4; ++kcp) {
            int kc0 = 2 * kcp, kc1 = 2 * kcp + 1;
            const _Float16* a0p = arow + (kc0 * 16 + m) * 32 + hi * 16;
            const _Float16* a1p = arow + (kc1 * 16 + m) * 32 + hi * 16;
            v8h a0l = *(const v8h*)a0p, a0h = *(const v8h*)(a0p + 8);
            v8h a1l = *(const v8h*)a1p, a1h = *(const v8h*)(a1p + 8);
            v16h a0, a1;
#pragma unroll
            for (int i = 0; i < 8; ++i) {
                a0[i] = a0l[i]; a0[i + 8] = a0h[i];
                a1[i] = a1l[i]; a1[i + 8] = a1h[i];
            }
            v16h b0 = *(const v16h*)(oWt + (size_t)(n0 + m) * 256 + kc0 * 32 + hi * 16);
            v16h b1 = *(const v16h*)(oWt + (size_t)(n0 + m) * 256 + kc1 * 32 + hi * 16);
            ca = __builtin_amdgcn_wmma_f32_16x16x32_f16(false, a0, false, b0,
                                                        (short)0, ca, false, false);
            cb = __builtin_amdgcn_wmma_f32_16x16x32_f16(false, a1, false, b1,
                                                        (short)0, cb, false, false);
        }
        v8f acc = ca + cb;
        int o = n0 + m;
        float bv = oB[o];
#pragma unroll
        for (int r = 0; r < 8; ++r) {
            int l = q0 + r + hi * 8;
            out[(size_t)(b * L_ + l) * O_ + o] = acc[r] + bv;
        }
    }
}

// ---------------------------------------------------------------------------
extern "C" void kernel_launch(void* const* d_in, const int* in_sizes, int n_in,
                              void* d_out, int out_size, void* d_ws, size_t ws_size,
                              hipStream_t stream) {
    const float* qin  = (const float*)d_in[0];
    const float* kvin = (const float*)d_in[1];
    const float* bias = (const float*)d_in[2];
    const float* qgW  = (const float*)d_in[3];
    const float* kvW  = (const float*)d_in[4];
    const float* qgB  = (const float*)d_in[5];
    const float* kvB  = (const float*)d_in[6];
    const float* oW   = (const float*)d_in[7];
    const float* oB   = (const float*)d_in[8];

    float* out      = (float*)d_out;                    // (B,L,O)
    float* attn_out = out + (size_t)B_ * L_ * O_;       // (B,Lq,Lk,H)

    char* ws = (char*)d_ws;
    const size_t MB = 1u << 20;
    _Float16* qf     = (_Float16*)(ws);                 // 2 MB
    _Float16* kf     = (_Float16*)(ws + 2 * MB);        // 2 MB
    _Float16* vT     = (_Float16*)(ws + 4 * MB);        // 2 MB
    float*    g32    = (float*)   (ws + 6 * MB);        // 4 MB
    _Float16* outg16 = (_Float16*)(ws + 10 * MB);       // 2 MB
    _Float16* qgWt   = (_Float16*)(ws + 12 * MB);       // 256 KB
    _Float16* kvWt   = (_Float16*)(ws + 12 * MB + 256 * 1024);
    _Float16* oWt    = (_Float16*)(ws + 12 * MB + 512 * 1024);   // 128 KB
    float*    qgBs   = (float*)   (ws + 12 * MB + 640 * 1024);   // 2 KB

    prepW_kernel<<<(256 * 512 + 255) / 256, 256, 0, stream>>>(qgW, qgWt, 256, 512, 1);
    prepW_kernel<<<(256 * 512 + 255) / 256, 256, 0, stream>>>(kvW, kvWt, 256, 512, 0);
    prepW_kernel<<<(256 * 256 + 255) / 256, 256, 0, stream>>>(oW,  oWt,  256, 256, 0);
    prepB_kernel<<<2, 256, 0, stream>>>(qgB, qgBs);

    projW_kernel<<<B_ * (L_ / 16), 128, 0, stream>>>(qin, kvin, qgWt, kvWt,
                                                     qgBs, kvB, qf, kf, vT, g32);

    size_t lds = (size_t)(2 * 16 * L_ + 256) * sizeof(float);   // 257 KB dynamic
    attn_kernel<<<B_ * H_ * (L_ / 16), 128, lds, stream>>>(
        qf, kf, vT, g32, bias, attn_out, outg16);

    outW_kernel<<<B_ * (L_ / 16), 128, 0, stream>>>(outg16, oWt, oB, out);
}
